// OneLayerTransformer_6777458393705
// MI455X (gfx1250) — matrix operand order repaired and checked by reading, not compiled, not run on hardware
//
#include <hip/hip_runtime.h>
#include <hip/hip_bf16.h>

// ---------------------------------------------------------------------------
// OneLayerTransformer for MI455X (gfx1250): persistent per-batch kernel.
//   grid = B = 32 workgroups (1 per batch, 1 per WGP), block = 256 (8 wave32).
//   Sliding-window state lives in LDS (~305 KB of the 320 KB/WGP).
//   Dominant op (561x561 mat-vec vs A_attn) on v_wmma_f32_16x16x32_bf16
//   against a pre-packed bf16 tile image of A_attn in d_ws.
// ---------------------------------------------------------------------------

typedef __attribute__((ext_vector_type(16))) __bf16 v16bf;
typedef __attribute__((ext_vector_type(8)))  float  v8f;

namespace olt {
constexpr int Bb   = 32;
constexpr int Tt   = 1024;
constexpr int NS   = 16;     // n_state
constexpr int NO   = 16;     // n_obs
constexpr int Hh   = 128;    // window
constexpr int Dd   = 32;     // NS + NO
constexpr int Ee   = 561;    // 1 + D + D*(D+1)/2
constexpr int CH4  = 71;     // float4s per score chunk (2*71*4 = 568 >= 561, zero tails)
constexpr int EPAD = 572;    // LDS row stride: 16B-aligned rows, 2-way b128 conflicts (floor)
constexpr int KT   = 18;     // ceil(576/32)  K tiles (bf16 k=32)
constexpr int NTP  = 40;     // N tiles padded to 8 waves * 5 tiles (640 >= 561)
constexpr int NQA  = NTP * 16;  // 640

constexpr size_t SMEM_FLOATS =
    (size_t)Hh * EPAD   // q_past rows (zero-padded cols 561..571)
  + 256                 // L_C
  + 256                 // A_LC
  + 576                 // q_query (padded, tail zero)
  + NQA                 // qA (padded, tail zero)
  + 256                 // score partials
  + 128                 // weights
  + 256                 // weighted-sum partials
  + 32                  // u_query (staged for NEXT step)
  + 32                  // u_new
  + 32;                 // u_weighted
constexpr size_t SMEM_BYTES = SMEM_FLOATS * 4 + 2 * 528 * sizeof(unsigned short);
} // namespace olt

// --- Prep: pack A_attn (561x561 f32, row-major [f][e]) into bf16 tiles in the
//     V_WMMA_16x16x32_BF16 B-operand lanes, contiguous k per half-wave (same order as A):
//     lane<16 : n = lane,    k = i      (i = 0..15)
//     lane>=16: n = lane-16, k = 16+i
//     tile (kt,nt): k = kt*32 + ..., n = nt*16 + ...   Zero padding outside 561.
__global__ void olt_pack_attn(const float* __restrict__ G, __bf16* __restrict__ P) {
    using namespace olt;
    const int NL = KT * NTP * 32;
    int gid = blockIdx.x * blockDim.x + threadIdx.x;
    if (gid >= NL) return;
    int kt   = gid / (NTP * 32);
    int r    = gid - kt * NTP * 32;
    int nt   = r >> 5;
    int lane = r & 31;
    int e     = nt * 16 + (lane & 15);
    int fbase = kt * 32 + ((lane < 16) ? 0 : 16);
#pragma unroll
    for (int i = 0; i < 16; ++i) {
        int f = fbase + i;
        float v = (f < Ee && e < Ee) ? G[f * Ee + e] : 0.0f;
        P[(size_t)gid * 16 + i] = (__bf16)v;
    }
}

__device__ __forceinline__ float olt_phi_elem(int e, const float* u,
                                              const unsigned short* qi,
                                              const unsigned short* qj) {
    if (e == 0) return 1.0f;
    if (e < 33) return u[e - 1];
    int q = e - 33;
    return u[qi[q]] * u[qj[q]];
}

__global__ void __launch_bounds__(256, 1)
olt_main(const float* __restrict__ y,    // [B,T,NO]
         const float* __restrict__ L,    // [NS,NO]
         const float* __restrict__ A,    // [NS,NS]
         const float* __restrict__ C,    // [NO,NS]
         const __bf16* __restrict__ gpk, // packed A_attn tiles
         float* __restrict__ out) {      // [B,T,NS]
    using namespace olt;
    extern __shared__ float smem[];
    float* s_qp  = smem;                      // [Hh][EPAD]
    float* s_LC  = s_qp + (size_t)Hh * EPAD;  // 256
    float* s_ALC = s_LC + 256;                // 256
    float* s_qq  = s_ALC + 256;               // 576
    float* s_qA  = s_qq + 576;                // 640
    float* s_sp  = s_qA + NQA;                // 256
    float* s_w   = s_sp + 256;                // 128
    float* s_pr  = s_w + 128;                 // 256
    float* s_uq  = s_pr + 256;                // 32  (query u for CURRENT step)
    float* s_un  = s_uq + 32;                 // 32
    float* s_uw  = s_un + 32;                 // 32
    unsigned short* s_qi = (unsigned short*)(s_uw + 32); // 528
    unsigned short* s_qj = s_qi + 528;                   // 528

    const int b    = blockIdx.x;
    const int tid  = threadIdx.x;
    const int lane = tid & 31;
    const int wave = tid >> 5;

    // ---------------- setup ----------------
    for (int q = tid; q < 528; q += 256) {          // triu index tables
        int i = 0, rem = q;
        while (rem >= Dd - i) { rem -= (Dd - i); ++i; }
        s_qi[q] = (unsigned short)i;
        s_qj[q] = (unsigned short)(i + rem);
    }
    {   // L_C = L@C ; A_LC = A - L_C  (one element per thread)
        int i = tid >> 4, j = tid & 15;
        float lc = 0.0f;
#pragma unroll
        for (int k = 0; k < NO; ++k) lc = fmaf(L[i * NO + k], C[k * NS + j], lc);
        s_LC[tid]  = lc;
        s_ALC[tid] = A[i * NS + j] - lc;
    }
    // zero q_past (stale garbage * 0-weight must not produce NaN; pad cols stay 0)
    for (int idx = tid; idx < (Hh * EPAD) / 4; idx += 256)
        ((float4*)s_qp)[idx] = float4{0.f, 0.f, 0.f, 0.f};
    for (int idx = tid; idx < 576; idx += 256) s_qq[idx] = 0.0f;
    for (int idx = tid; idx < NQA; idx += 256) s_qA[idx] = 0.0f;
    __syncthreads();

    // t = 0: x0 = L @ y[:,0]; window slot 0 <- phi([x0, y0]); stage query for t=1
    if (tid < NS) {
        float v = 0.0f;
#pragma unroll
        for (int j = 0; j < NO; ++j) v = fmaf(L[tid * NO + j], y[((size_t)b * Tt) * NO + j], v);
        out[((size_t)b * Tt) * NS + tid] = v;
        s_un[tid] = v;
        s_uq[tid] = v;                                   // x0
    } else if (tid < Dd) {
        s_un[tid] = y[((size_t)b * Tt) * NO + (tid - NS)];       // y_0
        s_uq[tid] = y[((size_t)b * Tt + 1) * NO + (tid - NS)];   // y_1
    }
    __syncthreads();
    for (int e = tid; e < Ee; e += 256) s_qp[e] = olt_phi_elem(e, s_un, s_qi, s_qj);
    __syncthreads();

    // ---------------- sequential steps ----------------
    for (int t = 1; t < Tt; ++t) {
        // (A) q_query = phi(u_query)     [s_uq staged by previous step / setup]
        for (int e = tid; e < Ee; e += 256) s_qq[e] = olt_phi_elem(e, s_uq, s_qi, s_qj);
        __syncthreads();

        // (B) qA = q_query @ A_attn via WMMA bf16 (row 0 of 16x576 A operand)
        {
            const int m  = lane & 15;
            const bool lo = lane < 16;
            v8f acc0{}, acc1{}, acc2{}, acc3{}, acc4{};
            const v16bf* __restrict__ gB = (const v16bf*)gpk;
            for (int kt = 0; kt < KT; ++kt) {
                // k-pattern = one contiguous 16-float run per half-wave -> b128 loads.
                const float4* q4 = (const float4*)(s_qq + kt * 32);
                float4 ra = q4[lo ? 0 : 4];   // k 0..3   / 16..19
                float4 rb = q4[lo ? 1 : 5];   // k 4..7   / 20..23
                float4 rc = q4[lo ? 2 : 6];   // k 8..11  / 24..27
                float4 rd = q4[lo ? 3 : 7];   // k 12..15 / 28..31
                float fa[16] = {ra.x, ra.y, ra.z, ra.w, rb.x, rb.y, rb.z, rb.w,
                                rc.x, rc.y, rc.z, rc.w, rd.x, rd.y, rd.z, rd.w};
                v16bf a;
#pragma unroll
                for (int i = 0; i < 16; ++i)
                    a[i] = (m == 0) ? (__bf16)fa[i] : (__bf16)0.0f;

                const size_t base = ((size_t)kt * NTP) * 32;
                v16bf b0 = gB[base + (size_t)(wave     ) * 32 + lane];
                acc0 = __builtin_amdgcn_wmma_f32_16x16x32_bf16(false, a, false, b0, (short)0, acc0, false, false);
                v16bf b1 = gB[base + (size_t)(wave +  8) * 32 + lane];
                acc1 = __builtin_amdgcn_wmma_f32_16x16x32_bf16(false, a, false, b1, (short)0, acc1, false, false);
                v16bf b2 = gB[base + (size_t)(wave + 16) * 32 + lane];
                acc2 = __builtin_amdgcn_wmma_f32_16x16x32_bf16(false, a, false, b2, (short)0, acc2, false, false);
                v16bf b3 = gB[base + (size_t)(wave + 24) * 32 + lane];
                acc3 = __builtin_amdgcn_wmma_f32_16x16x32_bf16(false, a, false, b3, (short)0, acc3, false, false);
                v16bf b4 = gB[base + (size_t)(wave + 32) * 32 + lane];
                acc4 = __builtin_amdgcn_wmma_f32_16x16x32_bf16(false, a, false, b4, (short)0, acc4, false, false);
            }
            if (lane < 16) {  // D row M=0 lives in element 0, lanes 0..15
                s_qA[(wave     ) * 16 + lane] = acc0[0];
                s_qA[(wave +  8) * 16 + lane] = acc1[0];
                s_qA[(wave + 16) * 16 + lane] = acc2[0];
                s_qA[(wave + 24) * 16 + lane] = acc3[0];
                s_qA[(wave + 32) * 16 + lane] = acc4[0];
            }
        }
        __syncthreads();

        // (C) scores_h partials: q_past[h] . qA
        //     Uniform code path: both chunks run an identical fixed 71-float4 trip
        //     (covers e in [0,568); cols 561..567 are zero on both operands), so the
        //     loop fully unrolls with no exec masking and LDS loads batch ahead.
        {
            int h = tid & 127, chunk = tid >> 7;
            const float4* row4 = (const float4*)(s_qp + (size_t)h * EPAD) + chunk * CH4;
            const float4* qa4  = (const float4*)s_qA + chunk * CH4;
            float ax = 0.f, ay = 0.f, az = 0.f, aw = 0.f;
#pragma unroll
            for (int e4 = 0; e4 < CH4; ++e4) {
                float4 r = row4[e4], q = qa4[e4];
                ax = fmaf(r.x, q.x, ax); ay = fmaf(r.y, q.y, ay);
                az = fmaf(r.z, q.z, az); aw = fmaf(r.w, q.w, aw);
            }
            s_sp[tid] = (ax + ay) + (az + aw);
        }
        __syncthreads();

        // (D) softmax entirely in wave 0 (mask -> max -> exp -> sum -> normalize)
        if (wave == 0) {
            float v[4];
#pragma unroll
            for (int k = 0; k < 4; ++k) {
                int h  = lane + 32 * k;
                int dh = ((t - 1) - h) & (Hh - 1);
                int s  = (t - 1) - dh;                  // timestep held by slot h
                float x = s_sp[h] + s_sp[h + 128];
                v[k] = (s >= 0) ? x : -3.0e38f;
            }
            float m = fmaxf(fmaxf(v[0], v[1]), fmaxf(v[2], v[3]));
#pragma unroll
            for (int off = 16; off > 0; off >>= 1) m = fmaxf(m, __shfl_xor(m, off));
            float e[4], sum = 0.f;
#pragma unroll
            for (int k = 0; k < 4; ++k) {
                e[k] = (v[k] > -1.0e37f) ? __expf(v[k] - m) : 0.0f;
                sum += e[k];
            }
#pragma unroll
            for (int off = 16; off > 0; off >>= 1) sum += __shfl_xor(sum, off);
            float inv = 1.0f / sum;
#pragma unroll
            for (int k = 0; k < 4; ++k) s_w[lane + 32 * k] = e[k] * inv;  // normalized
        }
        __syncthreads();

        // (E) uw partials: sum_h w_h * u_h   (u stored in phi positions 1..32)
        {
            int c = tid & 31, grp = tid >> 5;
            int h0 = grp * 16;
            float p = 0.0f;
#pragma unroll
            for (int hh = 0; hh < 16; ++hh)
                p = fmaf(s_w[h0 + hh], s_qp[(size_t)(h0 + hh) * EPAD + 1 + c], p);
            s_pr[tid] = p;
        }
        __syncthreads();

        // (F) wave 0: uw reduce -> x_new -> emit -> stage u_new & next u_query
        if (wave == 0) {
            float s = 0.0f;
#pragma unroll
            for (int g = 0; g < 8; ++g) s += s_pr[g * 32 + lane];
            s_uw[lane] = s;                    // weights already normalized
            // intra-wave LDS ordering: no block barrier needed before reading s_uw
            if (lane < NS) {
                float v = 0.0f;
#pragma unroll
                for (int j = 0; j < NS; ++j)
                    v = fmaf(s_ALC[lane * NS + j], s_uw[j],
                             fmaf(s_LC[lane * NS + j], s_uw[NS + j], v));
                out[((size_t)b * Tt + t) * NS + lane] = v;
                s_un[lane] = v;                // x_t
                s_uq[lane] = v;                // query x for step t+1
            } else {
                s_un[lane] = s_uq[lane];       // y_t (staged last step)
                int tn = (t + 1 < Tt) ? (t + 1) : (Tt - 1);
                s_uq[lane] = y[((size_t)b * Tt + tn) * NO + (lane - NS)];  // y_{t+1}
            }
        }
        __syncthreads();

        // (G) append window row: q_past[t & 127] = phi([x_t, y_t])
        {
            float* dst = s_qp + (size_t)(t & (Hh - 1)) * EPAD;
            for (int e = tid; e < Ee; e += 256) dst[e] = olt_phi_elem(e, s_un, s_qi, s_qj);
        }
        __syncthreads();
    }
}

extern "C" void kernel_launch(void* const* d_in, const int* in_sizes, int n_in,
                              void* d_out, int out_size, void* d_ws, size_t ws_size,
                              hipStream_t stream) {
    (void)in_sizes; (void)n_in; (void)out_size; (void)ws_size;
    const float* y = (const float*)d_in[0];
    const float* L = (const float*)d_in[1];
    const float* A = (const float*)d_in[2];
    const float* C = (const float*)d_in[3];
    const float* G = (const float*)d_in[4];   // A_attn
    float* out = (float*)d_out;
    __bf16* P  = (__bf16*)d_ws;               // 18*40*32*16 bf16 = 720 KB

    // allow >64KB dynamic LDS (CDNA5 WGP has 320 KB)
    static_cast<void>(hipFuncSetAttribute(
        reinterpret_cast<const void*>(&olt_main),
        hipFuncAttributeMaxDynamicSharedMemorySize, (int)olt::SMEM_BYTES));

    const int NL = olt::KT * olt::NTP * 32;
    olt_pack_attn<<<(NL + 255) / 256, 256, 0, stream>>>(G, P);
    olt_main<<<olt::Bb, 256, olt::SMEM_BYTES, stream>>>(y, L, A, C, P, out);
}
